// RestormerBlock_14413910245923
// MI455X (gfx1250) — matrix operand order repaired and checked by reading, not compiled, not run on hardware
//
#include <hip/hip_runtime.h>
#include <hip/hip_bf16.h>
#include <math.h>

// ---------------------------------------------------------------------------
// Restormer block on MI455X (gfx1250).  All GEMMs via v_wmma_f32_16x16x32_bf16
// (wave32, f32 accumulate).  bf16 activations everywhere except the residual
// stream.  GEMMs: one block per 64-column slab, B tile staged once in LDS
// (transposed -> per-lane contiguous ds_load_b128 fragments), all M-tiles
// computed from that slab.  Depthwise conv stages its row tile into LDS with
// GLOBAL_LOAD_ASYNC_TO_LDS_B128 + s_wait_asynccnt (CDNA5 async path).
// Workspace requirement: ~551 MB.
// ---------------------------------------------------------------------------

typedef __bf16 bf16;
typedef __attribute__((ext_vector_type(16))) __bf16 v16bf;
typedef __attribute__((ext_vector_type(8)))  float  v8f;

constexpr int LSP = 31 * 128 * 128;   // 507904 spatial positions, /64 and /8192 exact
constexpr int C   = 48;

// ---------------- weight convert + zero-pad (f32 -> bf16, K padded) ---------
__global__ void k_cvt(const float* __restrict__ src, bf16* __restrict__ dst,
                      int M, int K, int Kpad) {
    int i = blockIdx.x * 256 + threadIdx.x;      // grid sized to Mpad*Kpad/256
    int m = i / Kpad, k = i % Kpad;
    float v = (m < M && k < K) ? src[m * K + k] : 0.f;
    dst[i] = (bf16)v;
}

// ---------------- LayerNorm over 48 channels (per spatial position) ---------
__global__ void k_ln(const float* __restrict__ x, const float* __restrict__ w,
                     const float* __restrict__ b, bf16* __restrict__ out) {
    size_t p = (size_t)blockIdx.x * 256 + threadIdx.x;
    float v[C];
    float s = 0.f, ss = 0.f;
#pragma unroll
    for (int c = 0; c < C; ++c) {
        float t = x[(size_t)c * LSP + p];
        v[c] = t; s += t; ss += t * t;
    }
    float mu  = s * (1.f / C);
    float var = ss * (1.f / C) - mu * mu;
    float inv = rsqrtf(var + 1e-6f);
#pragma unroll
    for (int c = 0; c < C; ++c)
        out[(size_t)c * LSP + p] = (bf16)((v[c] - mu) * inv * w[c] + b[c]);
}

// ---------------- WMMA GEMM:  C[MREAL,L] = A[MREAL,KPAD] * B[KREAL,L] -------
// Block = 128 threads (4 waves) -> one 64-column slab of N, ALL M tiles.
// B slab staged once into LDS, transposed: Bs[col][k]; per-lane B fragment is
// then one contiguous 32-byte LDS read (2x ds_load_b128).  K pad rows zeroed
// in LDS, so there are no bounds checks in the inner loop (all compile-time).
// MODE 0: store bf16.   MODE 1: store f32 = C + resid (residual fused).
template <int MT, int KPAD, int KREAL, int MREAL, int MODE>
__global__ void k_gemm(const bf16* __restrict__ A, const bf16* __restrict__ B,
                       bf16* __restrict__ outB, float* __restrict__ outF,
                       const float* __restrict__ resid) {
    __shared__ __align__(32) bf16 Bs[64 * KPAD];      // [col][k], transposed
    const int tid  = threadIdx.x;
    const int lane = tid & 31;
    const int wv   = tid >> 5;
    const size_t nbase = (size_t)blockIdx.x * 64;

    // ---- stage B slab (coalesced b128 global loads, transposed ds stores) --
#pragma unroll
    for (int f = tid; f < KPAD * 8; f += 128) {       // KPAD*64 elems / 8
        int k = f >> 3, g = (f & 7) * 8;
        if (k < KREAL) {
            const bf16* src = B + (size_t)k * LSP + nbase + g;
#pragma unroll
            for (int i = 0; i < 8; ++i) Bs[(g + i) * KPAD + k] = src[i];
        } else {
#pragma unroll
            for (int i = 0; i < 8; ++i) Bs[(g + i) * KPAD + k] = (bf16)0.f;
        }
    }
    __syncthreads();

    const int col  = lane & 15;
    const int half = lane >> 4;
    v8f acc[MT] = {};
#pragma unroll
    for (int k0 = 0; k0 < KPAD; k0 += 32) {
        // B fragment: 16 contiguous bf16 per lane from transposed LDS tile
        v16bf bf = *(const v16bf*)&Bs[(wv * 16 + col) * KPAD + k0 + half * 16];
#pragma unroll
        for (int mt = 0; mt < MT; ++mt) {
            const bf16* ar = A + (size_t)(mt * 16 + col) * KPAD + k0;
            v16bf a;
#pragma unroll
            for (int j = 0; j < 8; ++j) {             // A 16x32: contiguous k = half*16 + s (matches B)
                int kb = half * 16 + 2 * j;
                a[2 * j]     = ar[kb];
                a[2 * j + 1] = ar[kb + 1];
            }
            acc[mt] = __builtin_amdgcn_wmma_f32_16x16x32_bf16(
                false, a, false, bf, (short)0, acc[mt], false, false);
        }
    }

    // ---- epilogue ----------------------------------------------------------
#pragma unroll
    for (int mt = 0; mt < MT; ++mt) {
#pragma unroll
        for (int r = 0; r < 8; ++r) {                 // C/D: vgpr r -> M=r+half*8
            int m = mt * 16 + r + half * 8;
            if (m >= MREAL) continue;                 // compile-time resolvable
            size_t idx = (size_t)m * LSP + nbase + wv * 16 + col;
            if (MODE == 0) outB[idx] = (bf16)acc[mt][r];
            else           outF[idx] = acc[mt][r] + resid[idx];
        }
    }
}

// ---------------- depthwise 3x3, async-LDS staged (CDNA5 path) --------------
// Block = 256 threads -> 2 output rows of one (channel, depth) slice.
// Rows y0-1..y0+2 (1 KB) staged via GLOBAL_LOAD_ASYNC_TO_LDS_B128 (ASYNCcnt);
// out-of-range halo rows zero-filled through the DS path by complementary
// lanes (no LDS overlap with async destinations -> single barrier suffices).
__global__ void k_dwconv(const bf16* __restrict__ in, const float* __restrict__ wt,
                         bf16* __restrict__ out) {
    __shared__ __align__(16) bf16 tile[4 * 128];      // rows y0-1 .. y0+2
    const int tid = threadIdx.x;
    const int p0  = blockIdx.x * 256;
    const int ch  = blockIdx.y;
    const int d   = p0 >> 14;
    const int y0  = (p0 & 16383) >> 7;                // even; block = rows y0,y0+1
    const bf16* src = in + (size_t)ch * LSP + (size_t)d * 16384;

    if (tid < 64) {                                   // 64 lanes x 16B = 1 KB tile
        int rr  = tid >> 4;                           // 0..3 -> rows y0-1..y0+2
        int ry  = y0 - 1 + rr;
        int seg = (tid & 15) * 8;                     // 8 bf16 = 16 bytes
        bf16* dst = tile + rr * 128 + seg;
        if (ry >= 0 && ry <= 127) {
            unsigned lds = (unsigned)(size_t)dst;     // LDS byte offset
            unsigned long long ga =
                (unsigned long long)(size_t)(src + ry * 128 + seg);
            asm volatile("global_load_async_to_lds_b128 %0, %1, off"
                         :: "v"(lds), "v"(ga) : "memory");
        } else {
            *(uint4*)dst = make_uint4(0u, 0u, 0u, 0u);
        }
    }
    asm volatile("s_wait_asynccnt 0x0" ::: "memory");
    __syncthreads();

    const float* w9 = wt + ch * 9;
    const int ly = (tid >> 7) + 1;                    // tile row of this output
    const int x  = tid & 127;
    float acc = 0.f;
#pragma unroll
    for (int dy = -1; dy <= 1; ++dy)
#pragma unroll
        for (int dx = -1; dx <= 1; ++dx) {
            int xx = x + dx;
            if (xx < 0 || xx > 127) continue;
            acc += w9[(dy + 1) * 3 + (dx + 1)] * (float)tile[(ly + dy) * 128 + xx];
        }
    out[(size_t)ch * LSP + p0 + tid] = (bf16)acc;
}

// depthwise 3x3 on both gate halves + exact GELU gating, fused (plain path)
__device__ inline float dwtap(const bf16* __restrict__ src,
                              const float* __restrict__ w9, int y, int x) {
    float acc = 0.f;
#pragma unroll
    for (int dy = -1; dy <= 1; ++dy) {
        int yy = y + dy; if (yy < 0 || yy > 127) continue;
#pragma unroll
        for (int dx = -1; dx <= 1; ++dx) {
            int xx = x + dx; if (xx < 0 || xx > 127) continue;
            acc += w9[(dy + 1) * 3 + (dx + 1)] * (float)src[yy * 128 + xx];
        }
    }
    return acc;
}

__global__ void k_dwgate(const bf16* __restrict__ H, const float* __restrict__ wt,
                         bf16* __restrict__ G) {
    int p  = blockIdx.x * 256 + threadIdx.x;
    int c1 = blockIdx.y;            // 0..126
    int c2 = c1 + 127;
    int d = p >> 14, yx = p & 16383, y = yx >> 7, x = yx & 127;
    const bf16* s1 = H + (size_t)c1 * LSP + (size_t)d * 16384;
    const bf16* s2 = H + (size_t)c2 * LSP + (size_t)d * 16384;
    __builtin_prefetch(s1 + yx + 16384, 0, 1);        // global_prefetch_b8
    float a  = dwtap(s1, wt + c1 * 9, y, x);
    float bv = dwtap(s2, wt + c2 * 9, y, x);
    float g  = 0.5f * a * (1.f + erff(a * 0.7071067811865475f));
    G[(size_t)c1 * LSP + p] = (bf16)(g * bv);
}

// ---------------- Gram = Q * K^T (48x48), split-K over L, WMMA + atomics ----
__global__ void k_gram(const bf16* __restrict__ qkv, float* __restrict__ G48) {
    int lane = threadIdx.x;
    int mt = blockIdx.x % 3, nt = blockIdx.x / 3;
    int base = blockIdx.y * 8192;
    const bf16* Q = qkv;
    const bf16* K = qkv + (size_t)C * LSP;
    int half = lane >> 4;
    int row  = mt * 16 + (lane & 15);
    int col  = nt * 16 + (lane & 15);
    v8f c = {};
    for (int kt = 0; kt < 256; ++kt) {
        int l0 = base + kt * 32;
        v16bf a, b;
#pragma unroll
        for (int j = 0; j < 8; ++j) {
            int kb = half * 16 + 2 * j;
            a[2 * j]     = Q[(size_t)row * LSP + l0 + kb];
            a[2 * j + 1] = Q[(size_t)row * LSP + l0 + kb + 1];
        }
#pragma unroll
        for (int j = 0; j < 8; ++j) {           // B(l,c) = K[c*L + l]
            int kb = half * 16 + 2 * j;
            b[2 * j]     = K[(size_t)col * LSP + l0 + kb];
            b[2 * j + 1] = K[(size_t)col * LSP + l0 + kb + 1];
        }
        c = __builtin_amdgcn_wmma_f32_16x16x32_bf16(false, a, false, b,
                                                    (short)0, c, false, false);
    }
#pragma unroll
    for (int r = 0; r < 8; ++r) {
        int m = mt * 16 + r + half * 8;
        int n = nt * 16 + (lane & 15);
        atomicAdd(&G48[m * C + n], c[r]);
    }
}

// ---------------- squared L2 norms of q,k rows over L -----------------------
__global__ void k_norms(const bf16* __restrict__ qkv, float* __restrict__ nq2,
                        float* __restrict__ nk2) {
    __shared__ float sq[256], sk[256];
    int ch = blockIdx.x, tid = threadIdx.x;
    const bf16* q = qkv + (size_t)ch * LSP;
    const bf16* k = qkv + (size_t)(C + ch) * LSP;
    int base = blockIdx.y * 8192 + tid;
    float aq = 0.f, ak = 0.f;
#pragma unroll 4
    for (int i = 0; i < 32; ++i) {
        float a = (float)q[base + i * 256]; aq += a * a;
        float b = (float)k[base + i * 256]; ak += b * b;
    }
    sq[tid] = aq; sk[tid] = ak; __syncthreads();
    for (int s = 128; s > 0; s >>= 1) {
        if (tid < s) { sq[tid] += sq[tid + s]; sk[tid] += sk[tid + s]; }
        __syncthreads();
    }
    if (tid == 0) { atomicAdd(&nq2[ch], sq[0]); atomicAdd(&nk2[ch], sk[0]); }
}

// ------- tiny: normalize, softmax, M = PO * blockdiag(attn) -> bf16[48x64] --
__global__ void k_attn(const float* __restrict__ G48, const float* __restrict__ nq2,
                       const float* __restrict__ nk2, const float* __restrict__ temp,
                       const float* __restrict__ po, bf16* __restrict__ Mbf) {
    __shared__ float nq[C], nk[C];
    __shared__ float attn[4][12][12];
    int tid = threadIdx.x;
    if (tid < C) {
        nq[tid] = fmaxf(sqrtf(nq2[tid]), 1e-12f);
        nk[tid] = fmaxf(sqrtf(nk2[tid]), 1e-12f);
    }
    __syncthreads();
    for (int i = tid; i < 576; i += 256) {
        int h = i / 144, r = (i / 12) % 12, kk = i % 12;
        int cg = h * 12 + r, kg = h * 12 + kk;
        attn[h][r][kk] = G48[cg * C + kg] / (nq[cg] * nk[kg]) * temp[h];
    }
    __syncthreads();
    if (tid < C) {                               // softmax over 12 keys
        int h = tid / 12, r = tid % 12;
        float mx = -1e30f;
        for (int kk = 0; kk < 12; ++kk) mx = fmaxf(mx, attn[h][r][kk]);
        float e[12], s = 0.f;
        for (int kk = 0; kk < 12; ++kk) { e[kk] = expf(attn[h][r][kk] - mx); s += e[kk]; }
        float inv = 1.f / s;
        for (int kk = 0; kk < 12; ++kk) attn[h][r][kk] = e[kk] * inv;
    }
    __syncthreads();
    for (int i = tid; i < C * 64; i += 256) {    // M = PO * A_blockdiag, K-padded
        int o = i >> 6, j = i & 63;
        float v = 0.f;
        if (j < C) {
            int h = j / 12, kk = j % 12;
            for (int r = 0; r < 12; ++r) v += po[o * C + h * 12 + r] * attn[h][r][kk];
        }
        Mbf[o * 64 + j] = (bf16)v;
    }
}

// ---------------------------------------------------------------------------
extern "C" void kernel_launch(void* const* d_in, const int* in_sizes, int n_in,
                              void* d_out, int out_size, void* d_ws, size_t ws_size,
                              hipStream_t stream) {
    const float* x      = (const float*)d_in[0];
    const float* n1w    = (const float*)d_in[1];
    const float* n1b    = (const float*)d_in[2];
    const float* qkv_w  = (const float*)d_in[3];
    const float* qkv_dw = (const float*)d_in[4];
    const float* temp   = (const float*)d_in[5];
    const float* po_w   = (const float*)d_in[6];
    const float* n2w    = (const float*)d_in[7];
    const float* n2b    = (const float*)d_in[8];
    const float* fin_w  = (const float*)d_in[9];
    const float* fdw    = (const float*)d_in[10];
    const float* fout_w = (const float*)d_in[11];
    float* out = (float*)d_out;
    char*  ws  = (char*)d_ws;

    size_t o = 0;
    auto take = [&](size_t bytes) { size_t r = o; o += (bytes + 255) & ~(size_t)255; return r; };
    size_t o_r0 = take((size_t)254 * LSP * 2);   // qkv_raw, later FFN hidden (aliased)
    size_t o_r1 = take((size_t)144 * LSP * 2);   // qkv post-dw, later gate (aliased)
    size_t o_xn = take((size_t)C * LSP * 2);     // LN output (reused for LN2)
    size_t o_x2 = take((size_t)C * LSP * 4);     // residual stream after MDTA (f32)
    size_t o_wq = take(144 * 64 * 2);
    size_t o_wi = take(256 * 64 * 2);
    size_t o_wo = take(48 * 128 * 2);
    size_t o_M  = take(48 * 64 * 2);
    size_t o_G  = take(48 * 48 * 4);
    size_t o_nq = take(48 * 4);
    size_t o_nk = take(48 * 4);
    (void)ws_size; (void)in_sizes; (void)n_in; (void)out_size;

    bf16*  qkvraw = (bf16*)(ws + o_r0);
    bf16*  Hraw   = (bf16*)(ws + o_r0);
    bf16*  qkv    = (bf16*)(ws + o_r1);
    bf16*  Ggate  = (bf16*)(ws + o_r1);
    bf16*  Xn     = (bf16*)(ws + o_xn);
    float* x2     = (float*)(ws + o_x2);
    bf16*  Wq     = (bf16*)(ws + o_wq);
    bf16*  Wi     = (bf16*)(ws + o_wi);
    bf16*  Wo     = (bf16*)(ws + o_wo);
    bf16*  Mbf    = (bf16*)(ws + o_M);
    float* G48    = (float*)(ws + o_G);
    float* nq2    = (float*)(ws + o_nq);
    float* nk2    = (float*)(ws + o_nk);

    hipMemsetAsync(G48, 0, 48 * 48 * 4, stream);
    hipMemsetAsync(nq2, 0, 48 * 4, stream);
    hipMemsetAsync(nk2, 0, 48 * 4, stream);

    dim3 b256(256), b128(128), b32(32);
    // weight prep (bf16, K zero-padded to 32-multiple, M to 16-multiple)
    k_cvt<<<36, b256, 0, stream>>>(qkv_w,  Wq, 144, 48, 64);
    k_cvt<<<64, b256, 0, stream>>>(fin_w,  Wi, 254, 48, 64);
    k_cvt<<<24, b256, 0, stream>>>(fout_w, Wo, 48, 127, 128);

    // ---- MDTA branch -------------------------------------------------------
    k_ln<<<LSP / 256, b256, 0, stream>>>(x, n1w, n1b, Xn);
    k_gemm<9, 64, 48, 144, 0><<<LSP / 64, b128, 0, stream>>>(Wq, Xn, qkvraw,
                                                             nullptr, nullptr);
    k_dwconv<<<dim3(LSP / 256, 144), b256, 0, stream>>>(qkvraw, qkv_dw, qkv);
    k_gram <<<dim3(9, LSP / 8192), b32, 0, stream>>>(qkv, G48);
    k_norms<<<dim3(48, LSP / 8192), b256, 0, stream>>>(qkv, nq2, nk2);
    k_attn <<<1, b256, 0, stream>>>(G48, nq2, nk2, temp, po_w, Mbf);
    // y = (PO*A_blockdiag) * V  + x   (residual fused)
    k_gemm<3, 64, 48, 48, 1><<<LSP / 64, b128, 0, stream>>>(
        Mbf, qkv + (size_t)96 * LSP, nullptr, x2, x);

    // ---- GDFN branch -------------------------------------------------------
    k_ln<<<LSP / 256, b256, 0, stream>>>(x2, n2w, n2b, Xn);
    k_gemm<16, 64, 48, 254, 0><<<LSP / 64, b128, 0, stream>>>(Wi, Xn, Hraw,
                                                              nullptr, nullptr);
    k_dwgate<<<dim3(LSP / 256, 127), b256, 0, stream>>>(Hraw, fdw, Ggate);
    k_gemm<3, 128, 127, 48, 1><<<LSP / 64, b128, 0, stream>>>(
        Wo, Ggate, nullptr, out, x2);
}
